// SAGE_35562329210946
// MI455X (gfx1250) — compile-verified, matrix operand order checked by reading, not run on hardware
//
#include <hip/hip_runtime.h>
#include <hip/hip_bf16.h>

#define NN 50000
#define NE 640000
#define F  128
#define KK 256   // concat dim for both layers

typedef _Float16 v16h __attribute__((ext_vector_type(16)));
typedef _Float16 v8h  __attribute__((ext_vector_type(8)));
typedef float    v8f  __attribute__((ext_vector_type(8)));

// ---------- degree: count in-edges per node (atomic f32 add into L2) ----------
__global__ void sage_deg_count(const int* __restrict__ dst, float* __restrict__ deg) {
    int e = blockIdx.x * blockDim.x + threadIdx.x;
    if (e < NE) {
        __hip_atomic_fetch_add(&deg[dst[e]], 1.0f, __ATOMIC_RELAXED, __HIP_MEMORY_SCOPE_AGENT);
    }
}

// ---------- invdeg[i] = 1 / max(deg,1), in place ----------
__global__ void sage_invdeg(float* __restrict__ deg) {
    int i = blockIdx.x * blockDim.x + threadIdx.x;
    if (i < NN) deg[i] = 1.0f / fmaxf(deg[i], 1.0f);
}

// ---------- edge-parallel scatter-add: agg[dst] += h[src]  (one wave per edge,
// each lane moves a float4 -> fully coalesced 512B per edge, atomics land in L2) ----------
__global__ void sage_aggregate(const float* __restrict__ h, const int* __restrict__ src,
                               const int* __restrict__ dst, float* __restrict__ agg) {
    long long tid = (long long)blockIdx.x * blockDim.x + threadIdx.x;
    int e = (int)(tid >> 5);
    if (e >= NE) return;
    int f = ((int)tid & 31) * 4;              // 32 lanes * 4 floats = 128 feats
    int s = src[e], d = dst[e];
    float4 v = *(const float4*)(h + (size_t)s * F + f);
    float* o = agg + (size_t)d * F + f;
    __hip_atomic_fetch_add(o + 0, v.x, __ATOMIC_RELAXED, __HIP_MEMORY_SCOPE_AGENT);
    __hip_atomic_fetch_add(o + 1, v.y, __ATOMIC_RELAXED, __HIP_MEMORY_SCOPE_AGENT);
    __hip_atomic_fetch_add(o + 2, v.z, __ATOMIC_RELAXED, __HIP_MEMORY_SCOPE_AGENT);
    __hip_atomic_fetch_add(o + 3, v.w, __ATOMIC_RELAXED, __HIP_MEMORY_SCOPE_AGENT);
}

// ---------- xcat[n] = f16( [ h[n] , agg[n]*invdeg[n] ] )  -> [NN, 256] f16 ----------
__global__ void sage_build_xcat(const float* __restrict__ h, const float* __restrict__ agg,
                                const float* __restrict__ invdeg, _Float16* __restrict__ xcat) {
    int tid = blockIdx.x * blockDim.x + threadIdx.x;
    if (tid >= NN * F) return;
    int n = tid >> 7, f = tid & 127;
    xcat[(size_t)n * KK + f]       = (_Float16)h[tid];
    xcat[(size_t)n * KK + F + f]   = (_Float16)(agg[tid] * invdeg[n]);
}

// ---------- pack W[k][n] (f32, row-major [256 x ncols]) into per-lane WMMA B-fragment
// order:  wp[(((tileN*8 + kstep)*32 + lane)*16) + j]   so the GEMM does contiguous b128 loads.
// 16-bit fragment K packing (ISA 7.12.2): half = lane>>4;
//   v=0..3 -> K = half*8 + 2v + {0,1};  v=4..7 -> K = 16 + half*8 + 2(v-4) + {0,1} ----------
__global__ void sage_pack_w(const float* __restrict__ W, _Float16* __restrict__ wp, int ncols) {
    int tid = blockIdx.x * blockDim.x + threadIdx.x;
    int total = (ncols / 16) * 8 * 32 * 16;
    if (tid >= total) return;
    int j     = tid & 15;
    int lane  = (tid >> 4) & 31;
    int kstep = (tid >> 9) & 7;
    int tileN = tid >> 12;
    int half  = lane >> 4;
    int v     = j >> 1;
    int k = kstep * 32 + ((v < 4) ? (half * 8 + v * 2) : (16 + half * 8 + (v - 4) * 2)) + (j & 1);
    int n = tileN * 16 + (lane & 15);
    wp[tid] = (_Float16)W[(size_t)k * ncols + n];
}

// ---------- out[tileM*16 + m, :] = relu?( xcat @ W + b )   one 16x16 tile per wave,
// 8 x v_wmma_f32_16x16x32_f16 over K=256, fused bias + optional ReLU ----------
__global__ void sage_gemm(const _Float16* __restrict__ xcat, const _Float16* __restrict__ wp,
                          const float* __restrict__ bias, float* __restrict__ out,
                          int ncols, int relu) {
    int wave  = threadIdx.x >> 5;     // = tileN
    int lane  = threadIdx.x & 31;
    int half  = lane >> 4;
    int mrow  = lane & 15;
    int tileM = blockIdx.x;

    const _Float16* arow = xcat + (size_t)(tileM * 16 + mrow) * KK;
    v8f c = {};
#pragma unroll
    for (int ks = 0; ks < 8; ++ks) {
        int kb = ks * 32;
        v8h a0 = *(const v8h*)(arow + kb + half * 8);
        v8h a1 = *(const v8h*)(arow + kb + 16 + half * 8);
        v16h a = __builtin_shufflevector(a0, a1, 0,1,2,3,4,5,6,7,8,9,10,11,12,13,14,15);

        const _Float16* bptr = wp + ((size_t)((wave * 8 + ks) * 32 + lane) << 4);
        v8h b0 = *(const v8h*)(bptr);
        v8h b1 = *(const v8h*)(bptr + 8);
        v16h b = __builtin_shufflevector(b0, b1, 0,1,2,3,4,5,6,7,8,9,10,11,12,13,14,15);

        c = __builtin_amdgcn_wmma_f32_16x16x32_f16(false, a, false, b, (short)0, c, false, false);
    }

    int col = wave * 16 + mrow;
    float bv = bias[col];
#pragma unroll
    for (int r = 0; r < 8; ++r) {
        int m = r + 8 * half;                       // D layout: VGPR r -> row r (+8 for hi lanes)
        float v = c[r] + bv;
        if (relu) v = fmaxf(v, 0.0f);
        out[(size_t)(tileM * 16 + m) * ncols + col] = v;
    }
}

extern "C" void kernel_launch(void* const* d_in, const int* in_sizes, int n_in,
                              void* d_out, int out_size, void* d_ws, size_t ws_size,
                              hipStream_t stream) {
    const float* features = (const float*)d_in[0];
    const int*   esrc     = (const int*)d_in[1];
    const int*   edst     = (const int*)d_in[2];
    const float* W1       = (const float*)d_in[3];
    const float* b1       = (const float*)d_in[4];
    const float* W2       = (const float*)d_in[5];
    const float* b2       = (const float*)d_in[6];
    (void)in_sizes; (void)n_in; (void)out_size; (void)ws_size;

    char* ws = (char*)d_ws;
    size_t off = 0;
    auto carve = [&](size_t bytes) { size_t p = off; off = (off + bytes + 255) & ~(size_t)255; return p; };
    float*    agg    = (float*)   (ws + carve((size_t)NN * F * 4));   // 25.6 MB
    float*    invdeg = (float*)   (ws + carve((size_t)NN * 4));       // 0.2 MB
    _Float16* xcat   = (_Float16*)(ws + carve((size_t)NN * KK * 2));  // 25.6 MB
    float*    h1     = (float*)   (ws + carve((size_t)NN * F * 4));   // 25.6 MB
    _Float16* wpack  = (_Float16*)(ws + carve((size_t)8 * 8 * 32 * 16 * 2)); // 64 KB

    // ---- shared pre-pass: degrees (edge_dst only, reused by both layers) ----
    hipMemsetAsync(invdeg, 0, (size_t)NN * 4, stream);
    hipMemsetAsync(agg,    0, (size_t)NN * F * 4, stream);
    sage_deg_count<<<(NE + 255) / 256, 256, 0, stream>>>(edst, invdeg);
    sage_invdeg  <<<(NN + 255) / 256, 256, 0, stream>>>(invdeg);

    // ---- layer 1: mean-agg(features) -> concat -> f16 WMMA GEMM + bias + ReLU ----
    sage_aggregate <<<(NE * 32) / 256, 256, 0, stream>>>(features, esrc, edst, agg);
    sage_build_xcat<<<(NN * F + 255) / 256, 256, 0, stream>>>(features, agg, invdeg, xcat);
    sage_pack_w    <<<(8 * 8 * 32 * 16 + 255) / 256, 256, 0, stream>>>(W1, wpack, 128);
    sage_gemm      <<<NN / 16, 256, 0, stream>>>(xcat, wpack, b1, h1, 128, 1);

    // ---- layer 2: mean-agg(h1) -> concat -> f16 WMMA GEMM + bias ----
    hipMemsetAsync(agg, 0, (size_t)NN * F * 4, stream);
    sage_aggregate <<<(NE * 32) / 256, 256, 0, stream>>>(h1, esrc, edst, agg);
    sage_build_xcat<<<(NN * F + 255) / 256, 256, 0, stream>>>(h1, agg, invdeg, xcat);
    sage_pack_w    <<<(4 * 8 * 32 * 16 + 255) / 256, 256, 0, stream>>>(W2, wpack, 64);
    sage_gemm      <<<NN / 16, 128, 0, stream>>>(xcat, wpack, b2, (float*)d_out, 64, 0);
}
